// ContrastivePredictiveCoding_78967268704393
// MI455X (gfx1250) — matrix operand order repaired and checked by reading, not compiled, not run on hardware
//
#include <hip/hip_runtime.h>
#include <hip/hip_bf16.h>

typedef _Float16 half_t;
typedef __attribute__((ext_vector_type(16))) _Float16 v16h;
typedef __attribute__((ext_vector_type(8)))  _Float16 v8h;
typedef __attribute__((ext_vector_type(8)))  float    v8f;
typedef __attribute__((ext_vector_type(4)))  float    v4f;

#define DIM 256
#define WAVES_PER_WG 4
#define ACT_STRIDE 264   // halves per activation-tile row (pad: stride 132 dwords)
#define ZB_STRIDE  264   // halves per staged z-tile row
#define LOG2E 1.4426950408889634f
#define LN2   0.6931471805599453f

// native transcendental ops: v_exp_f32 (2^x) and v_log_f32 (log2 x)
__device__ __forceinline__ float fast_exp2(float x) { return __builtin_amdgcn_exp2f(x); }
__device__ __forceinline__ float fast_log2(float x) { return __builtin_amdgcn_logf(x); }

// ---------------------------------------------------------------------------
// helpers
// ---------------------------------------------------------------------------
__device__ __forceinline__ v16h combine8(v8h lo, v8h hi) {
  v16h r;
#pragma unroll
  for (int i = 0; i < 8; ++i) { r[i] = lo[i]; r[8 + i] = hi[i]; }
  return r;
}

// Reload the wave's 16x256 A-operand registers from its LDS activation tile.
// lane: m = lane&15, h = lane>>4.  A[c][e]=X[m][32c+8h+e] (e<8), +16 for e>=8.
__device__ __forceinline__ void reload_a(const half_t* act, int m, int h, v16h A[8]) {
#pragma unroll
  for (int c = 0; c < 8; ++c) {
    v8h lo = *(const v8h*)(act + m * ACT_STRIDE + 32 * c + 8 * h);
    v8h hi = *(const v8h*)(act + m * ACT_STRIDE + 32 * c + 16 + 8 * h);
    A[c] = combine8(lo, hi);
  }
}

// K=256 accumulation. Wt is [N][256] f16 row-major, so the B-operand
// (k, n=lane&15) for chunk c = Wt[col][32c+8h .. +8] and Wt[col][32c+16+8h .. +8]: two 16B loads.
__device__ __forceinline__ v8f wmma_k256(const v16h a[8], const half_t* __restrict__ Wt,
                                         int col, int h, v8f acc) {
  const half_t* wr = Wt + (size_t)col * DIM;
#pragma unroll
  for (int c = 0; c < 8; ++c) {
    v16h b = combine8(*(const v8h*)(wr + 32 * c + 8 * h), *(const v8h*)(wr + 32 * c + 16 + 8 * h));
    acc = __builtin_amdgcn_wmma_f32_16x16x32_f16(false, a[c], false, b,
                                                 (short)0, acc, false, false);
  }
  return acc;
}

__device__ __forceinline__ float sigmoidf_(float x) { return 1.0f / (1.0f + __expf(-x)); }
__device__ __forceinline__ float tanhf_(float x)    { return 2.0f / (1.0f + __expf(-2.0f * x)) - 1.0f; }

// Dense layer in place: A <- act( A @ W + b ).  Output goes C-layout -> LDS
// activation tile; A registers are refilled once at the end (keeps peak
// register pressure at ~A(64)+acc(16)+B pipeline: no scratch spills).
template <bool RELU>
__device__ __forceinline__ void dense_act(v16h A[8],
                                          const half_t* __restrict__ Wt,
                                          const float* __restrict__ bias,
                                          half_t* act, int lane) {
  const int n = lane & 15, h = lane >> 4, m = n;
#pragma unroll 1
  for (int t = 0; t < 8; ++t) {
#pragma unroll
    for (int sub = 0; sub < 2; ++sub) {
      const int col = (2 * t + sub) * 16 + n;
      v8f acc = {};
      acc = wmma_k256(A, Wt, col, h, acc);
      const float bv = bias[col];
#pragma unroll
      for (int r = 0; r < 8; ++r) {
        float v = acc[r] + bv;
        if (RELU) v = fmaxf(v, 0.0f);
        act[(r + 8 * h) * ACT_STRIDE + col] = (half_t)v;
      }
    }
  }
  reload_a(act, m, h, A);   // wave-local LDS: HW keeps DS ops in-order per wave
}

// GRU step with h0 = 0 (torch gate order r,z,n): A <- (1-z) * tanh(gi_n + r*bhh_n)
__device__ __forceinline__ void gru_act(v16h A[8],
                                        const half_t* __restrict__ Wiht,
                                        const float* __restrict__ bih,
                                        const float* __restrict__ bhh,
                                        half_t* act, int lane) {
  const int n = lane & 15, h = lane >> 4, m = n;
#pragma unroll 1
  for (int t = 0; t < 8; ++t) {
#pragma unroll 1                       // keep only one sub's 24 accumulators live
    for (int sub = 0; sub < 2; ++sub) {
      const int col = (2 * t + sub) * 16 + n;
      v8f ar = {}, az = {}, an = {};
      ar = wmma_k256(A, Wiht, col, h, ar);
      az = wmma_k256(A, Wiht, col + DIM, h, az);
      an = wmma_k256(A, Wiht, col + 2 * DIM, h, an);
      const float br  = bih[col] + bhh[col];
      const float bz  = bih[col + DIM] + bhh[col + DIM];
      const float bn  = bih[col + 2 * DIM];
      const float bhn = bhh[col + 2 * DIM];
#pragma unroll
      for (int r = 0; r < 8; ++r) {
        float rg = sigmoidf_(ar[r] + br);
        float zg = sigmoidf_(az[r] + bz);
        float ng = tanhf_(an[r] + bn + rg * bhn);
        act[(r + 8 * h) * ACT_STRIDE + col] = (half_t)((1.0f - zg) * ng);
      }
    }
  }
  reload_a(act, m, h, A);
}

// ---------------------------------------------------------------------------
// kernel 1: weight convert + transpose  W[K][N] f32 -> Wt[N][K] f16
// ---------------------------------------------------------------------------
__global__ void cpc_convert_wt(const float* __restrict__ W, half_t* __restrict__ Wt,
                               int K, int N) {
  int idx = blockIdx.x * blockDim.x + threadIdx.x;
  if (idx >= K * N) return;
  int k = idx / N, n = idx - k * N;
  Wt[(size_t)n * K + k] = (half_t)W[idx];
}

// ---------------------------------------------------------------------------
// kernel 2: fused  encoder [-> GRU -> predictor].  1 wave = 16 batch rows.
// mode 0: out = encoder(x)          (z_tp1 path)
// mode 1: out = pred(gru(encoder))  (pred path)
// ---------------------------------------------------------------------------
__global__ __launch_bounds__(WAVES_PER_WG * 32)
void cpc_fused(const float* __restrict__ x,
               const half_t* __restrict__ wt1, const float* __restrict__ b1,
               const half_t* __restrict__ wt2, const float* __restrict__ b2,
               const half_t* __restrict__ wiht, const float* __restrict__ bih,
               const float* __restrict__ bhh,
               const half_t* __restrict__ pt1, const float* __restrict__ pb1,
               const half_t* __restrict__ pt2, const float* __restrict__ pb2,
               half_t* __restrict__ out, int mode) {
  __shared__ half_t act_all[WAVES_PER_WG * 16 * ACT_STRIDE];
  const int wave = threadIdx.x >> 5;
  const int lane = threadIdx.x & 31;
  half_t* act = act_all + wave * 16 * ACT_STRIDE;
  const int row0 = (blockIdx.x * WAVES_PER_WG + wave) * 16;
  const int m = lane & 15, h = lane >> 4;

  v16h A[8];

  // load input tile (f32 -> f16) straight into WMMA A-layout registers
  {
    const float* rowp = x + (size_t)(row0 + m) * DIM;
#pragma unroll
    for (int c = 0; c < 8; ++c) {
      const float* p0 = rowp + 32 * c + 8 * h;
      const float* p1 = rowp + 32 * c + 16 + 8 * h;
      v4f q0 = *(const v4f*)p0, q1 = *(const v4f*)(p0 + 4);
      v4f q2 = *(const v4f*)p1, q3 = *(const v4f*)(p1 + 4);
      v16h a;
#pragma unroll
      for (int i = 0; i < 4; ++i) {
        a[i] = (half_t)q0[i]; a[4 + i] = (half_t)q1[i];
        a[8 + i] = (half_t)q2[i]; a[12 + i] = (half_t)q3[i];
      }
      A[c] = a;
    }
  }

  dense_act<true >(A, wt1, b1, act, lane);
  dense_act<false>(A, wt2, b2, act, lane);

  if (mode != 0) {
    gru_act(A, wiht, bih, bhh, act, lane);
    dense_act<true >(A, pt1, pb1, act, lane);
    dense_act<false>(A, pt2, pb2, act, lane);
  }

  // store result (f16 row-major) from A-layout: two 16B contiguous stores/chunk
  half_t* rowp = out + (size_t)(row0 + m) * DIM;
#pragma unroll
  for (int c = 0; c < 8; ++c) {
    v8h lo, hi;
#pragma unroll
    for (int i = 0; i < 8; ++i) { lo[i] = A[c][i]; hi[i] = A[c][8 + i]; }
    *(v8h*)(rowp + 32 * c + 8 * h)      = lo;
    *(v8h*)(rowp + 32 * c + 16 + 8 * h) = hi;
  }
}

// ---------------------------------------------------------------------------
// kernel 3: fused InfoNCE logits + online log-softmax row stats.
// Per row i: rowout[i] = logits[i,i] - logsumexp_j(logits[i,j])
// Each lane keeps a private base-2 online (max,sum) over its own column
// subset (j = n mod 16) -> zero in-loop shuffles; single 16-lane merge at end.
// Workgroup stages 32 z_tp1 rows per barrier round, shared by 4 waves.
// ---------------------------------------------------------------------------
__global__ __launch_bounds__(WAVES_PER_WG * 32)
void cpc_logits(const half_t* __restrict__ pred, const half_t* __restrict__ z1,
                float* __restrict__ rowout, int Btot) {
  __shared__ half_t zb[32 * ZB_STRIDE];
  const int wave = threadIdx.x >> 5;
  const int lane = threadIdx.x & 31;
  const int row0 = (blockIdx.x * WAVES_PER_WG + wave) * 16;
  const int n = lane & 15, h = lane >> 4, m = n;

  // wave's pred rows, A-layout, held in registers for the whole j sweep
  v16h A[8];
  {
    const half_t* rowp = pred + (size_t)(row0 + m) * DIM;
#pragma unroll
    for (int c = 0; c < 8; ++c) {
      v8h lo = *(const v8h*)(rowp + 32 * c + 8 * h);
      v8h hi = *(const v8h*)(rowp + 32 * c + 16 + 8 * h);
      A[c] = combine8(lo, hi);
    }
  }

  float M[8], S[8], DG[8];               // per-lane stats, base-2 domain
#pragma unroll
  for (int r = 0; r < 8; ++r) { M[r] = -1.0e30f; S[r] = 0.0f; DG[r] = 0.0f; }

  const float sc = 10.0f * LOG2E;        // (1/TEMPERATURE) * log2(e)

  for (int j0 = 0; j0 < Btot; j0 += 32) {
    __syncthreads();
    {  // cooperative stage of z1[j0..j0+31][:] (128 thr x 128B)
      const int t = threadIdx.x;
      const int zr = t >> 2, seg = t & 3;
      const half_t* src = z1 + (size_t)(j0 + zr) * DIM + seg * 64;
      half_t* dst = zb + zr * ZB_STRIDE + seg * 64;
#pragma unroll
      for (int q = 0; q < 4; ++q)
        *(v16h*)(dst + 16 * q) = *(const v16h*)(src + 16 * q);
    }
    __syncthreads();

#pragma unroll
    for (int sub = 0; sub < 2; ++sub) {
      const half_t* bp = zb + (16 * sub + n) * ZB_STRIDE;  // B col n = z1 row
      v8f acc = {};
#pragma unroll
      for (int c = 0; c < 8; ++c) {
        v16h b = combine8(*(const v8h*)(bp + 32 * c + 8 * h), *(const v8h*)(bp + 32 * c + 16 + 8 * h));
        acc = __builtin_amdgcn_wmma_f32_16x16x32_f16(false, A[c], false, b,
                                                     (short)0, acc, false, false);
      }
      const int jcol = j0 + 16 * sub + n;
#pragma unroll
      for (int r = 0; r < 8; ++r) {
        float u = acc[r] * sc;                       // logit in log2 domain
        if (jcol == row0 + r + 8 * h) DG[r] = u;     // diagonal element
        float nm = fmaxf(M[r], u);
        S[r] = S[r] * fast_exp2(M[r] - nm) + fast_exp2(u - nm);
        M[r] = nm;
      }
    }
  }

  // merge the 16 per-lane (max,sum) pairs and the diagonal; write row result
#pragma unroll
  for (int r = 0; r < 8; ++r) {
    float nm = M[r];
#pragma unroll
    for (int off = 1; off < 16; off <<= 1)
      nm = fmaxf(nm, __shfl_xor(nm, off, 16));
    float e = S[r] * fast_exp2(M[r] - nm);
#pragma unroll
    for (int off = 1; off < 16; off <<= 1)
      e += __shfl_xor(e, off, 16);
    float dv = DG[r];
#pragma unroll
    for (int off = 1; off < 16; off <<= 1)
      dv += __shfl_xor(dv, off, 16);
    if (n == 0)
      rowout[row0 + r + 8 * h] = LN2 * (dv - nm - fast_log2(e));
  }
}

// ---------------------------------------------------------------------------
// kernel 4: loss = -mean(rowout)
// ---------------------------------------------------------------------------
__global__ void cpc_reduce(const float* __restrict__ rowout, float* __restrict__ out,
                           int Btot) {
  __shared__ float sm[256];
  float s = 0.0f;
  for (int i = threadIdx.x; i < Btot; i += blockDim.x) s += rowout[i];
  sm[threadIdx.x] = s;
  __syncthreads();
  for (int k = 128; k > 0; k >>= 1) {
    if ((int)threadIdx.x < k) sm[threadIdx.x] += sm[threadIdx.x + k];
    __syncthreads();
  }
  if (threadIdx.x == 0) out[0] = -sm[0] / (float)Btot;
}

// ---------------------------------------------------------------------------
// launch
// ---------------------------------------------------------------------------
extern "C" void kernel_launch(void* const* d_in, const int* in_sizes, int n_in,
                              void* d_out, int out_size, void* d_ws, size_t ws_size,
                              hipStream_t stream) {
  const float* state      = (const float*)d_in[0];
  /* d_in[1] action: unused by the loss */
  const float* next_state = (const float*)d_in[2];
  const float* enc_w1 = (const float*)d_in[3];
  const float* enc_b1 = (const float*)d_in[4];
  const float* enc_w2 = (const float*)d_in[5];
  const float* enc_b2 = (const float*)d_in[6];
  const float* gru_wih = (const float*)d_in[7];
  /* d_in[8] gru_whh: dead (h0 == 0) */
  const float* gru_bih = (const float*)d_in[9];
  const float* gru_bhh = (const float*)d_in[10];
  const float* pred_w1 = (const float*)d_in[11];
  const float* pred_b1 = (const float*)d_in[12];
  const float* pred_w2 = (const float*)d_in[13];
  const float* pred_b2 = (const float*)d_in[14];

  const int Btot = in_sizes[0] / DIM;   // 8192

  char* ws = (char*)d_ws;
  size_t off = 0;
  auto alloc = [&](size_t bytes) -> void* {
    void* p = ws + off;
    off += (bytes + 255) & ~(size_t)255;
    return p;
  };
  half_t* wt1   = (half_t*)alloc((size_t)DIM * DIM * 2);
  half_t* wt2   = (half_t*)alloc((size_t)DIM * DIM * 2);
  half_t* wiht  = (half_t*)alloc((size_t)3 * DIM * DIM * 2);
  half_t* pt1   = (half_t*)alloc((size_t)DIM * DIM * 2);
  half_t* pt2   = (half_t*)alloc((size_t)DIM * DIM * 2);
  half_t* z1    = (half_t*)alloc((size_t)Btot * DIM * 2);
  half_t* predh = (half_t*)alloc((size_t)Btot * DIM * 2);
  float*  rowst = (float*)alloc((size_t)Btot * 4);

  // 1) weights -> transposed f16
  {
    int n1 = DIM * DIM, n3 = DIM * 3 * DIM;
    cpc_convert_wt<<<(n1 + 255) / 256, 256, 0, stream>>>(enc_w1, wt1, DIM, DIM);
    cpc_convert_wt<<<(n1 + 255) / 256, 256, 0, stream>>>(enc_w2, wt2, DIM, DIM);
    cpc_convert_wt<<<(n3 + 255) / 256, 256, 0, stream>>>(gru_wih, wiht, DIM, 3 * DIM);
    cpc_convert_wt<<<(n1 + 255) / 256, 256, 0, stream>>>(pred_w1, pt1, DIM, DIM);
    cpc_convert_wt<<<(n1 + 255) / 256, 256, 0, stream>>>(pred_w2, pt2, DIM, DIM);
  }

  const int blocks = Btot / (16 * WAVES_PER_WG);   // 128
  // 2) z_tp1 = encoder(next_state)
  cpc_fused<<<blocks, WAVES_PER_WG * 32, 0, stream>>>(
      next_state, wt1, enc_b1, wt2, enc_b2, wiht, gru_bih, gru_bhh,
      pt1, pred_b1, pt2, pred_b2, z1, 0);
  // 3) pred = predictor(gru(encoder(state)))
  cpc_fused<<<blocks, WAVES_PER_WG * 32, 0, stream>>>(
      state, wt1, enc_b1, wt2, enc_b2, wiht, gru_bih, gru_bhh,
      pt1, pred_b1, pt2, pred_b2, predh, 1);
  // 4) fused logits + online log-softmax diagonal
  cpc_logits<<<blocks, WAVES_PER_WG * 32, 0, stream>>>(predh, z1, rowst, Btot);
  // 5) final scalar loss
  cpc_reduce<<<1, 256, 0, stream>>>(rowst, (float*)d_out, Btot);
}
